// JointGlobalPointer_51496657879542
// MI455X (gfx1250) — compile-verified, matrix operand order checked by reading, not run on hardware
//
#include <hip/hip_runtime.h>

// Problem dims
#define B_   8
#define L_   1024
#define D_   1024          // BERT_DIM
#define H_   12            // HEADS
#define HD_  64            // HEAD_SIZE
#define N_   (H_ * HD_)    // 768
#define M_   (B_ * L_)     // 8192

// LDS staging for weight tiles in the projection GEMM (double buffered)
#define KCH        128      // K-chunk staged in LDS (bf16 elems per row)
#define LDS_ROW_DW 68       // 64 dwords data + 4 dwords pad -> bank offset 4/row

typedef __attribute__((ext_vector_type(16))) __bf16        v16bf;
typedef __attribute__((ext_vector_type(8)))  float         v8f;
typedef __attribute__((ext_vector_type(4)))  unsigned int  u32x4;
typedef __attribute__((ext_vector_type(8)))  int           i32x8;
typedef __attribute__((ext_vector_type(4)))  int           i32x4;

#if defined(__has_builtin)
#  if __has_builtin(__builtin_amdgcn_tensor_load_to_lds)
#    define HAVE_TDM 1
#  else
#    define HAVE_TDM 0
#  endif
#else
#  define HAVE_TDM 0
#endif

union FragBF {
    u32x4 u[2];
    v16bf v;
};

// float -> bf16 (round to nearest even), branch free
static __device__ __forceinline__ unsigned short f2bf(float f) {
    unsigned int u = __float_as_uint(f);
    u += 0x7FFFu + ((u >> 16) & 1u);
    return (unsigned short)(u >> 16);
}

// Load a 16x32 bf16 A/B fragment from row-major global memory.
// Lane layout per CDNA5 ISA 16-bit A-matrix 16x32:
//   lanes 0-15 : K = k0+0..7   (VGPR0-3), K = k0+16..23 (VGPR4-7)
//   lanes 16-31: K = k0+8..15  (VGPR0-3), K = k0+24..31 (VGPR4-7)
// Caller pre-offsets the pointer by (row*ld + k0 + (lane&16 ? 8 : 0)).
static __device__ __forceinline__ v16bf ldfrag(const unsigned short* p) {
    FragBF f;
    f.u[0] = *(const u32x4*)(p);
    f.u[1] = *(const u32x4*)(p + 16);
    return f.v;
}

// Fragment from LDS-staged weight tile (padded rows of LDS_ROW_DW dwords).
static __device__ __forceinline__ v16bf ldsfrag(const unsigned int* lds,
                                                int row, int kc, int lane) {
    const unsigned int* p = lds + row * LDS_ROW_DW + (kc >> 1) + ((lane & 16) ? 4 : 0);
    FragBF f;
    f.u[0] = *(const u32x4*)(p);
    f.u[1] = *(const u32x4*)(p + 8);
    return f.v;
}

static __device__ __forceinline__ v8f wmma_bf16(v16bf a, v16bf b, v8f c) {
    return __builtin_amdgcn_wmma_f32_16x16x32_bf16(
        /*neg_a=*/false, a, /*neg_b=*/false, b,
        /*c_mod=*/(short)0, c, /*reuse_a=*/false, /*reuse_b=*/false);
}

#if HAVE_TDM
// Issue one TDM 2D tile load: 64 rows x KCH bf16 from row-major WT (row stride D_)
// into LDS at ldsAddr with padding: 4 dwords inserted every 64 dwords
// (pad_interval code 5, pad_amount code 3) -> LDS row stride = 68 dwords.
static __device__ __forceinline__ void tdm_load_tile(const unsigned short* gsrc,
                                                     unsigned ldsAddr) {
    unsigned long long ga = (unsigned long long)(uintptr_t)gsrc;
    u32x4 g0;
    g0[0] = 1u;                                   // count=1, user descriptor
    g0[1] = ldsAddr;                              // lds_addr (bytes)
    g0[2] = (unsigned)ga;                         // global_addr[31:0]
    g0[3] = (unsigned)((ga >> 32) & 0x01FFFFFFu)  // global_addr[56:32]
          | (2u << 30);                           // type = 2 ("image")
    i32x8 g1;
    g1[0] = (int)((1u << 16)        // data_size = 1 (2 bytes)
                | (1u << 20)        // pad_enable
                | (5u << 22)        // pad_interval code 5 = 64 dwords (one row)
                | (3u << 25));      // pad_amount  code 3 = 4 dwords
    g1[1] = (int)(((unsigned)D_ & 0xFFFFu) << 16);            // tensor_dim0 lo16
    g1[2] = (int)((((unsigned)D_ >> 16) & 0xFFFFu)
                | (((unsigned)N_ & 0xFFFFu) << 16));          // dim0 hi / dim1 lo
    g1[3] = (int)((((unsigned)N_ >> 16) & 0xFFFFu)
                | ((unsigned)KCH << 16));                     // dim1 hi / tile_dim0
    g1[4] = 64;                                               // tile_dim1 = 64 rows
    g1[5] = D_;                                               // tensor_dim0_stride lo32
    g1[6] = 0;
    g1[7] = 0;
    i32x4 gz4 = {0, 0, 0, 0};                                 // 2D: groups 2/3 unused
    i32x8 gz8 = {0, 0, 0, 0, 0, 0, 0, 0};                     // extra group (clang-23 form)
    __builtin_amdgcn_tensor_load_to_lds(g0, g1, gz4, gz4, gz8, 0);
}
#endif

// Cooperative fallback copy with the identical padded LDS layout.
static __device__ __forceinline__ void coop_copy_tile(const unsigned short* gB,
                                                      unsigned int* lds, int tid) {
    // 64 rows x 128 bf16 = 1024 units of 16B; 256 threads -> 4 units each.
    for (int u = tid; u < 64 * 16; u += 256) {
        int row = u >> 4;
        int c4  = u & 15;                      // 16B unit within row
        u32x4 srcv = *(const u32x4*)(gB + (size_t)row * D_ + c4 * 8);
        *(u32x4*)(lds + row * LDS_ROW_DW + c4 * 4) = srcv;
    }
}

// ---------------------------------------------------------------------------
// Stage 0a: fp32 -> bf16 conversion, 4 elements/thread
// ---------------------------------------------------------------------------
__global__ __launch_bounds__(256) void cvt_bf16_kernel(
    const float* __restrict__ in, unsigned short* __restrict__ out, int n4) {
    int i = blockIdx.x * blockDim.x + threadIdx.x;
    if (i >= n4) return;
    float4 v = ((const float4*)in)[i];
    uint2 r;
    r.x = (unsigned)f2bf(v.x) | ((unsigned)f2bf(v.y) << 16);
    r.y = (unsigned)f2bf(v.z) | ((unsigned)f2bf(v.w) << 16);
    ((uint2*)out)[i] = r;
}

// ---------------------------------------------------------------------------
// Stage 0b: W (D_ x N_) -> W^T bf16 (N_ x D_)
// ---------------------------------------------------------------------------
__global__ __launch_bounds__(256) void cvt_wT_kernel(
    const float* __restrict__ W, unsigned short* __restrict__ WT) {
    int i = blockIdx.x * blockDim.x + threadIdx.x;  // over N_ * D_
    int n = i >> 10;        // / D_
    int k = i & (D_ - 1);
    WT[i] = f2bf(W[(size_t)k * N_ + n]);
}

// ---------------------------------------------------------------------------
// Stage 0c: RoPE cos/sin tables, [L_][32] (one entry per rotation pair)
// ---------------------------------------------------------------------------
__global__ __launch_bounds__(256) void rope_tab_kernel(
    float* __restrict__ cosb, float* __restrict__ sinb) {
    int i = blockIdx.x * blockDim.x + threadIdx.x;  // L_ * 32
    int l = i >> 5;
    int j = i & 31;
    float inv = __powf(10000.0f, -2.0f * (float)j / (float)HD_);
    float ang = (float)l * inv;
    cosb[i] = __cosf(ang);
    sinb[i] = __sinf(ang);
}

// ---------------------------------------------------------------------------
// Stage 1: projection GEMM (bf16 WMMA, fp32 accum) + bias + RoPE.
// Block tile: 128 rows x 64 cols (one head). 8 waves, each 16 rows x 4 tiles.
// Weight K-chunks staged in LDS by the Tensor Data Mover, double buffered so
// the DMA overlaps WMMA work.  The K-step loop is software-pipelined in
// registers: fragments for step kc+32 are loaded before the WMMAs of step kc.
// ---------------------------------------------------------------------------
__global__ __launch_bounds__(256) void proj_rope_kernel(
    const unsigned short* __restrict__ Xb,    // M_ x D_ bf16
    const unsigned short* __restrict__ WT,    // N_ x D_ bf16 (transposed W)
    const float* __restrict__ bias,           // N_
    const float* __restrict__ cosb,           // L_ x 32
    const float* __restrict__ sinb,           // L_ x 32
    unsigned short* __restrict__ outp)        // B_ x H_ x L_ x HD_ bf16
{
    __shared__ unsigned int ldsB[2][64 * LDS_ROW_DW];   // 2 x 17408 bytes

    const int lane    = threadIdx.x & 31;
    const int w       = threadIdx.x >> 5;
    const int headIdx = blockIdx.x;               // 0..11 (N-tile == head)
    const int row0    = blockIdx.y * 128 + w * 16;
    const int n0      = headIdx * HD_;

    const int lm  = lane & 15;
    const int sel = (lane & 16) ? 8 : 0;

    const unsigned short* aB = Xb + (size_t)(row0 + lm) * D_ + sel;
    const unsigned short* gB = WT + (size_t)n0 * D_;

    v8f acc[4] = {};

    const int NC = D_ / KCH;                       // 8 chunks
    // Prologue: fill buffer 0 with chunk 0.
#if HAVE_TDM
    if (w == 0) {
        tdm_load_tile(gB, (unsigned)(uintptr_t)&ldsB[0][0]);
        __builtin_amdgcn_s_wait_tensorcnt(0);
    }
#else
    coop_copy_tile(gB, &ldsB[0][0], threadIdx.x);
#endif
    __syncthreads();

    for (int c = 0; c < NC; ++c) {
        const unsigned int* cur = &ldsB[c & 1][0];
        // Kick off the next chunk's DMA into the other buffer (overlaps compute).
        if (c + 1 < NC) {
#if HAVE_TDM
            if (w == 0)
                tdm_load_tile(gB + (c + 1) * KCH,
                              (unsigned)(uintptr_t)&ldsB[(c + 1) & 1][0]);
#else
            coop_copy_tile(gB + (c + 1) * KCH, &ldsB[(c + 1) & 1][0], threadIdx.x);
#endif
        }

        const unsigned short* aC = aB + c * KCH;

        // Software-pipelined K-steps: fragments held in distinct registers,
        // next step's loads issued before current step's WMMAs.
        v16bf a_cur = ldfrag(aC);
        v16bf b_cur[4];
#pragma unroll
        for (int t = 0; t < 4; ++t)
            b_cur[t] = ldsfrag(cur, t * 16 + lm, 0, lane);

#pragma unroll
        for (int kc = 32; kc < KCH; kc += 32) {
            v16bf a_nxt = ldfrag(aC + kc);
            v16bf b_nxt[4];
#pragma unroll
            for (int t = 0; t < 4; ++t)
                b_nxt[t] = ldsfrag(cur, t * 16 + lm, kc, lane);
            __builtin_prefetch(aC + kc + KCH, 0, 0);       // next chunk of A
#pragma unroll
            for (int t = 0; t < 4; ++t)
                acc[t] = wmma_bf16(a_cur, b_cur[t], acc[t]);
            a_cur = a_nxt;
#pragma unroll
            for (int t = 0; t < 4; ++t)
                b_cur[t] = b_nxt[t];
        }
#pragma unroll
        for (int t = 0; t < 4; ++t)
            acc[t] = wmma_bf16(a_cur, b_cur[t], acc[t]);

#if HAVE_TDM
        if (w == 0) __builtin_amdgcn_s_wait_tensorcnt(0);
#endif
        __syncthreads();
    }

    // Epilogue: bias + interleaved RoPE (pair partner lives in lane^1).
    // Even lanes pack both halves of a rotation pair into one b32 store.
    const int mhalf = (lane >> 4) << 3;   // C layout: lanes16-31 hold M = r+8
#pragma unroll
    for (int t = 0; t < 4; ++t) {
        const int d  = t * 16 + lm;       // 0..63 within head
        const int j  = d >> 1;            // rotation pair index
        const float bv = bias[n0 + d];
#pragma unroll
        for (int r = 0; r < 8; ++r) {
            const int Mg   = row0 + r + mhalf;
            const int bidx = Mg >> 10;          // / L_
            const int lpos = Mg & (L_ - 1);
            float val = acc[t][r] + bv;
            float partner = __shfl_xor(val, 1, 32);
            float cv = cosb[lpos * 32 + j];
            float sv = sinb[lpos * 32 + j];
            // even d: x*cos - x_pair*sin ; odd d: x*cos + x_pair*sin
            float o = (d & 1) ? (val * cv + partner * sv)
                              : (val * cv - partner * sv);
            float o2 = __shfl_xor(o, 1, 32);    // odd-lane result to even lane
            if (!(d & 1)) {
                unsigned pk = (unsigned)f2bf(o) | ((unsigned)f2bf(o2) << 16);
                size_t off = (((size_t)(bidx * H_ + headIdx)) * L_ + lpos) * HD_ + d;
                *(unsigned*)(outp + off) = pk;
            }
        }
    }
}

// ---------------------------------------------------------------------------
// Stage 2: per-(b,head) S = H * T^T, scaled by 1/sqrt(64) = 0.125.
// Block tile 128x128, 8 waves as 4x2, each wave 32 rows x 64 cols.
// K = 64: all 12 fragments for both K-steps are loaded before the 16 WMMAs,
// so there is a single load-wait instead of one per WMMA.  The kernel is
// bound by the 402 MB fp32 output stream (~17 us at 23.3 TB/s).
// ---------------------------------------------------------------------------
__global__ __launch_bounds__(256) void scores_kernel(
    const unsigned short* __restrict__ Hr,   // B_*H_ x L_ x HD_ bf16
    const unsigned short* __restrict__ Tr,   // B_*H_ x L_ x HD_ bf16
    float* __restrict__ out)                 // B_*H_ x L_ x L_ fp32
{
    const int lane = threadIdx.x & 31;
    const int w    = threadIdx.x >> 5;
    const int bh   = blockIdx.y;                 // 0..95
    const int tm   = blockIdx.x >> 3;            // 0..7 row block
    const int tn   = blockIdx.x & 7;             // 0..7 col block
    const int wr   = w & 3;                      // row group (32)
    const int wc   = w >> 2;                     // col group (64)
    const int m0   = tm * 128 + wr * 32;
    const int n0   = tn * 128 + wc * 64;

    const unsigned short* hB = Hr + (size_t)bh * L_ * HD_;
    const unsigned short* tB = Tr + (size_t)bh * L_ * HD_;
    const int lm  = lane & 15;
    const int sel = (lane & 16) ? 8 : 0;

    // Load every fragment for the whole K range first.
    v16bf a[2][2];     // [kstep][mtile]
    v16bf bq[2][4];    // [kstep][ntile]
#pragma unroll
    for (int ks = 0; ks < 2; ++ks) {
        const int k0 = ks * 32 + sel;
        a[ks][0] = ldfrag(hB + (size_t)(m0 + lm)      * HD_ + k0);
        a[ks][1] = ldfrag(hB + (size_t)(m0 + 16 + lm) * HD_ + k0);
#pragma unroll
        for (int t = 0; t < 4; ++t)
            bq[ks][t] = ldfrag(tB + (size_t)(n0 + t * 16 + lm) * HD_ + k0);
    }

    v8f acc[2][4] = {};
#pragma unroll
    for (int ks = 0; ks < 2; ++ks)
#pragma unroll
        for (int t = 0; t < 4; ++t) {
            acc[0][t] = wmma_bf16(a[ks][0], bq[ks][t], acc[0][t]);
            acc[1][t] = wmma_bf16(a[ks][1], bq[ks][t], acc[1][t]);
        }

    const int mhalf = (lane >> 4) << 3;
    float* ob = out + (size_t)bh * L_ * L_;
#pragma unroll
    for (int i = 0; i < 2; ++i)
#pragma unroll
        for (int t = 0; t < 4; ++t) {
            const int Ng = n0 + t * 16 + lm;
#pragma unroll
            for (int r = 0; r < 8; ++r) {
                const int Mg = m0 + i * 16 + r + mhalf;
                ob[(size_t)Mg * L_ + Ng] = acc[i][t][r] * 0.125f;
            }
        }
}

// ---------------------------------------------------------------------------
extern "C" void kernel_launch(void* const* d_in, const int* in_sizes, int n_in,
                              void* d_out, int out_size, void* d_ws, size_t ws_size,
                              hipStream_t stream) {
    (void)in_sizes; (void)n_in; (void)out_size; (void)ws_size;
    const float* head   = (const float*)d_in[0];
    const float* tail   = (const float*)d_in[1];
    const float* W_head = (const float*)d_in[2];
    const float* b_head = (const float*)d_in[3];
    const float* W_tail = (const float*)d_in[4];
    const float* b_tail = (const float*)d_in[5];
    float* out = (float*)d_out;

    // Workspace layout (bytes):
    char* ws = (char*)d_ws;
    unsigned short* headb = (unsigned short*)ws; ws += (size_t)M_ * D_ * 2;      // 16 MiB
    unsigned short* tailb = (unsigned short*)ws; ws += (size_t)M_ * D_ * 2;      // 16 MiB
    unsigned short* WhT   = (unsigned short*)ws; ws += (size_t)N_ * D_ * 2;      // 1.5 MiB
    unsigned short* WtT   = (unsigned short*)ws; ws += (size_t)N_ * D_ * 2;      // 1.5 MiB
    unsigned short* hrope = (unsigned short*)ws; ws += (size_t)B_ * H_ * L_ * HD_ * 2; // 12 MiB
    unsigned short* trope = (unsigned short*)ws; ws += (size_t)B_ * H_ * L_ * HD_ * 2; // 12 MiB
    float* cosb = (float*)ws; ws += (size_t)L_ * 32 * 4;
    float* sinb = (float*)ws; ws += (size_t)L_ * 32 * 4;

    const int nact4 = (M_ * D_) / 4;
    cvt_bf16_kernel<<<nact4 / 256, 256, 0, stream>>>(head, headb, nact4);
    cvt_bf16_kernel<<<nact4 / 256, 256, 0, stream>>>(tail, tailb, nact4);
    cvt_wT_kernel<<<(N_ * D_) / 256, 256, 0, stream>>>(W_head, WhT);
    cvt_wT_kernel<<<(N_ * D_) / 256, 256, 0, stream>>>(W_tail, WtT);
    rope_tab_kernel<<<(L_ * 32) / 256, 256, 0, stream>>>(cosb, sinb);

    dim3 pg(H_, M_ / 128);   // 12 x 64 blocks
    proj_rope_kernel<<<pg, 256, 0, stream>>>(headb, WhT, b_head, cosb, sinb, hrope);
    proj_rope_kernel<<<pg, 256, 0, stream>>>(tailb, WtT, b_tail, cosb, sinb, trope);

    dim3 sg(64, B_ * H_);    // 64 tiles x 96 batch-heads
    scores_kernel<<<sg, 256, 0, stream>>>(hrope, trope, out);
}
